// EdgeBlockONNX_53206054863196
// MI455X (gfx1250) — matrix operand order repaired and checked by reading, not compiled, not run on hardware
//
#include <hip/hip_runtime.h>

// ---------------------------------------------------------------------------
// EdgeBlock MLP for MI455X (gfx1250, wave32, WMMA bf16)
// out = LayerNorm( ReLU([sender||receiver||edge] @ W1 + b1) @ W2 + b2 )
// ---------------------------------------------------------------------------

typedef __attribute__((ext_vector_type(16))) __bf16        v16bf;
typedef __attribute__((ext_vector_type(8)))  float         v8f;
typedef __attribute__((ext_vector_type(4)))  unsigned int  u32x4;
typedef __attribute__((ext_vector_type(4)))  __bf16        bf16x4;

#define D      128
#define DIN    384
#define SW1    392   // padded K-stride for 384-wide bf16 rows (+16B -> no bank conflicts)
#define SW2    136   // padded K-stride for 128-wide bf16 rows
#define MBLK   128   // edges per workgroup iteration
#define NTHR   256   // 8 waves

union ABFrag { v16bf v; u32x4 u[2]; };

// A fragment: two runs of 8 consecutive K at +0 and +16 (ISA 16-bit A layout)
#define LDA(f, arr, idx)                                   \
  {                                                        \
    (f).u[0] = *(const u32x4*)&(arr)[(idx)];               \
    (f).u[1] = *(const u32x4*)&(arr)[(idx) + 16];          \
  }
// B fragment: two runs of 8 consecutive K at +0 and +16 (matches A layout)
#define LDB(f, arr, idx)                                   \
  {                                                        \
    (f).u[0] = *(const u32x4*)&(arr)[(idx)];               \
    (f).u[1] = *(const u32x4*)&(arr)[(idx) + 16];          \
  }

// --- kernel 0: transpose + f32->bf16 convert weights into workspace --------
__global__ void prep_weights(const float* __restrict__ W1,
                             const float* __restrict__ W2,
                             __bf16* __restrict__ w1t,
                             __bf16* __restrict__ w2t) {
  int i = blockIdx.x * NTHR + threadIdx.x;
  const int tot1 = D * SW1;
  const int tot2 = D * SW2;
  if (i < tot1) {
    int n = i / SW1;
    int k = i - n * SW1;
    float v = (k < DIN) ? W1[k * D + n] : 0.0f;
    w1t[i] = (__bf16)v;
  } else if (i < tot1 + tot2) {
    int j = i - tot1;
    int n = j / SW2;
    int k = j - n * SW2;
    float v = (k < D) ? W2[k * D + n] : 0.0f;
    w2t[j] = (__bf16)v;
  }
}

// --- main fused kernel ------------------------------------------------------
__global__ __launch_bounds__(NTHR)
void edge_mlp(const float* __restrict__ node_attr,
              const float* __restrict__ edge_attr,
              const long long* __restrict__ eidx,   // [2][E] int64
              const float* __restrict__ b1,
              const float* __restrict__ b2,
              const float* __restrict__ gammav,
              const float* __restrict__ betav,
              const __bf16* __restrict__ w1t,       // [128][SW1] bf16 (K-major)
              const __bf16* __restrict__ w2t,       // [128][SW2] bf16
              float* __restrict__ out,
              int E, int nBlocks) {
  __shared__ __align__(16) __bf16 w1t_s[D * SW1];    // 100,352 B
  __shared__ __align__(16) __bf16 w2t_s[D * SW2];    //  34,816 B
  __shared__ __align__(16) __bf16 x_s[MBLK * SW1];   // 100,352 B
  __shared__ __align__(16) __bf16 h_s[MBLK * SW2];   //  34,816 B
  __shared__ int sidx_s[MBLK];
  __shared__ int ridx_s[MBLK];

  const int tid  = threadIdx.x;
  const int lane = tid & 31;
  const int wm   = tid >> 5;      // wave id == M-strip (16 rows each)
  const int kh   = lane >> 4;     // half-wave id
  const int ln   = lane & 15;     // lane within half

  // stage weights to LDS once (persistent workgroup)
  {
    const u32x4* s1 = (const u32x4*)w1t;
    u32x4*       d1 = (u32x4*)w1t_s;
    for (int i = tid; i < D * SW1 / 8; i += NTHR) d1[i] = s1[i];
    const u32x4* s2 = (const u32x4*)w2t;
    u32x4*       d2 = (u32x4*)w2t_s;
    for (int i = tid; i < D * SW2 / 8; i += NTHR) d2[i] = s2[i];
  }

  // per-lane epilogue constants: column n = 16*t + ln
  float b1v[8], b2v[8], gv[8], btv[8];
#pragma unroll
  for (int t = 0; t < 8; ++t) {
    int n = t * 16 + ln;
    b1v[t] = b1[n]; b2v[t] = b2[n]; gv[t] = gammav[n]; btv[t] = betav[n];
  }

  const v8f zv = {0.f, 0.f, 0.f, 0.f, 0.f, 0.f, 0.f, 0.f};

  for (int blk = blockIdx.x; blk < nBlocks; blk += gridDim.x) {
    const int base = blk * MBLK;

    // --- load edge indices for this block ---
    if (tid < MBLK) {
      int eg = base + tid;
      long long s = 0, r = 0;
      if (eg < E) { s = eidx[eg]; r = eidx[(size_t)E + eg]; }
      sidx_s[tid] = (int)s;
      ridx_s[tid] = (int)r;
    }
    __syncthreads();

    // --- gather + convert x = [sender || receiver || edge] into LDS (bf16) ---
    // 128 edges * 3 segments * 32 float4 = 12288 float4 units
    for (int i = tid; i < MBLK * 96; i += NTHR) {
      int e   = i / 96;
      int rem = i - e * 96;
      int seg = rem >> 5;
      int off = rem & 31;
      int eg  = base + e;
      if (eg < E) {
        const float* src;
        if (seg == 0)      src = node_attr + (size_t)sidx_s[e] * D;
        else if (seg == 1) src = node_attr + (size_t)ridx_s[e] * D;
        else               src = edge_attr + (size_t)eg * D;
        float4 f = ((const float4*)src)[off];
        bf16x4 b;
        b.x = (__bf16)f.x; b.y = (__bf16)f.y; b.z = (__bf16)f.z; b.w = (__bf16)f.w;
        *(bf16x4*)&x_s[e * SW1 + (seg << 7) + (off << 2)] = b;
      }
    }
    __syncthreads();

    // --- prefetch next block's streamed data (edge_attr + indices) into L2 ---
    {
      int nb = blk + gridDim.x;
      if (nb < nBlocks) {
        int base2 = nb * MBLK;
        int rem   = E - base2; if (rem > MBLK) rem = MBLK;
        const float* ep = edge_attr + (size_t)base2 * D;
        for (int j = tid; j < rem * 4; j += NTHR)        // 128B cacheline steps
          __builtin_prefetch(ep + (size_t)j * 32, 0, 1);
        if (tid < 16) {
          __builtin_prefetch(&eidx[base2 + tid * 8], 0, 1);
          __builtin_prefetch(&eidx[(size_t)E + base2 + tid * 8], 0, 1);
        }
      }
    }

    // --- GEMM1: [128 x 384] @ [384 x 128], K = 12 steps of 32 ---
    // Software-pipelined: B double-buffered across N-tiles, A across K-steps.
    v8f acc[8];
#pragma unroll
    for (int t = 0; t < 8; ++t) acc[t] = zv;

    const int arow1 = (wm * 16 + ln) * SW1;
    const int brow1 = ln * SW1;
    {
      ABFrag a, an;
      LDA(a, x_s, arow1 + kh * 8);
#pragma unroll
      for (int ks = 0; ks < 12; ++ks) {
        if (ks < 11) { LDA(an, x_s, arow1 + (ks + 1) * 32 + kh * 8); }
        const int kb = ks * 32 + kh * 8;
        ABFrag b0, b1;
        LDB(b0, w1t_s, brow1 + kb);
#pragma unroll
        for (int t = 0; t < 8; ++t) {
          if (t < 7) { LDB(b1, w1t_s, brow1 + (t + 1) * 16 * SW1 + kb); }
          acc[t] = __builtin_amdgcn_wmma_f32_16x16x32_bf16(
              false, a.v, false, b0.v, (short)0, acc[t], false, false);
          b0 = b1;
        }
        a = an;
      }
    }

    // --- epilogue 1: bias + ReLU -> bf16 h tile in LDS ---
    // Rows written here (wm*16 .. wm*16+15) are read back only by this same
    // wave in GEMM2; same-wave LDS ops are in-order, so no barrier needed.
#pragma unroll
    for (int t = 0; t < 8; ++t) {
      int col = t * 16 + ln;
#pragma unroll
      for (int v = 0; v < 8; ++v) {
        float xv = acc[t][v] + b1v[t];
        xv = fmaxf(xv, 0.0f);
        int m = wm * 16 + kh * 8 + v;            // C layout: row = v + 8*half
        h_s[m * SW2 + col] = (__bf16)xv;
      }
    }

    // --- GEMM2: [128 x 128] @ [128 x 128], K = 4 steps of 32 ---
#pragma unroll
    for (int t = 0; t < 8; ++t) acc[t] = zv;

    const int arow2 = (wm * 16 + ln) * SW2;
    const int brow2 = ln * SW2;
    {
      ABFrag a, an;
      LDA(a, h_s, arow2 + kh * 8);
#pragma unroll
      for (int ks = 0; ks < 4; ++ks) {
        if (ks < 3) { LDA(an, h_s, arow2 + (ks + 1) * 32 + kh * 8); }
        const int kb = ks * 32 + kh * 8;
        ABFrag b0, b1;
        LDB(b0, w2t_s, brow2 + kb);
#pragma unroll
        for (int t = 0; t < 8; ++t) {
          if (t < 7) { LDB(b1, w2t_s, brow2 + (t + 1) * 16 * SW2 + kb); }
          acc[t] = __builtin_amdgcn_wmma_f32_16x16x32_bf16(
              false, a.v, false, b0.v, (short)0, acc[t], false, false);
          b0 = b1;
        }
        a = an;
      }
    }

    // --- epilogue 2: bias + in-register LayerNorm + store ---
    // Row r's 128 values live on the 16 lanes of one half-wave across 8 tiles,
    // at fixed accumulator element v. Reduce with 4 xor-shuffles (within half).
#pragma unroll
    for (int v = 0; v < 8; ++v) {
      float vals[8];
      float p = 0.0f, q = 0.0f;
#pragma unroll
      for (int t = 0; t < 8; ++t) {
        float xv = acc[t][v] + b2v[t];
        vals[t] = xv;
        p += xv;
        q += xv * xv;
      }
#pragma unroll
      for (int md = 1; md <= 8; md <<= 1) {
        p += __shfl_xor(p, md, 32);
        q += __shfl_xor(q, md, 32);
      }
      float mu  = p * (1.0f / 128.0f);
      float var = q * (1.0f / 128.0f) - mu * mu;
      float inv = rsqrtf(var + 1e-5f);
      int m  = wm * 16 + kh * 8 + v;
      int eg = base + m;
      if (eg < E) {
        float* orow = out + (size_t)eg * D;
#pragma unroll
        for (int t = 0; t < 8; ++t)
          orow[t * 16 + ln] = (vals[t] - mu) * inv * gv[t] + btv[t];
      }
    }
  }
}

// ---------------------------------------------------------------------------
extern "C" void kernel_launch(void* const* d_in, const int* in_sizes, int n_in,
                              void* d_out, int out_size, void* d_ws, size_t ws_size,
                              hipStream_t stream) {
  const float*     node_attr = (const float*)d_in[0];
  const float*     edge_attr = (const float*)d_in[1];
  const long long* eidx      = (const long long*)d_in[2];
  const float*     W1        = (const float*)d_in[3];
  const float*     b1        = (const float*)d_in[4];
  const float*     W2        = (const float*)d_in[5];
  const float*     b2        = (const float*)d_in[6];
  const float*     gamma_    = (const float*)d_in[7];
  const float*     beta_     = (const float*)d_in[8];
  float*           out       = (float*)d_out;

  const int E = in_sizes[1] / D;

  __bf16* w1t = (__bf16*)d_ws;                 // 128*SW1 bf16 = 100,352 B
  __bf16* w2t = w1t + D * SW1;                 // 128*SW2 bf16 =  34,816 B

  {
    int tot = D * SW1 + D * SW2;
    int g   = (tot + NTHR - 1) / NTHR;
    prep_weights<<<g, NTHR, 0, stream>>>(W1, W2, w1t, w2t);
  }

  int nBlocks = (E + MBLK - 1) / MBLK;
  int grid    = nBlocks < 1024 ? nBlocks : 1024;
  if (grid > 0) {
    edge_mlp<<<grid, NTHR, 0, stream>>>(node_attr, edge_attr, eidx,
                                        b1, b2, gamma_, beta_,
                                        w1t, w2t, out, E, nBlocks);
  }
}
